// RCNCell_22771916603454
// MI455X (gfx1250) — matrix operand order repaired and checked by reading, not compiled, not run on hardware
//
#include <hip/hip_runtime.h>
#include <hip/hip_bf16.h>

typedef long long i64;
typedef __attribute__((ext_vector_type(16))) __bf16 v16bf;
typedef __attribute__((ext_vector_type(8)))  float  v8f;
typedef __attribute__((ext_vector_type(4))) unsigned int u32x4;
typedef __attribute__((ext_vector_type(8))) int  i32x8;
typedef __attribute__((ext_vector_type(4))) int  i32x4;

#define Qv  32
#define Nn  8192
#define Hh  128
#define Dd  64
#define Rr  64
#define NHc (Nn * Hh)   /* 1048576 */
#define THc (3 * Hh)    /* 384 */

#if __has_builtin(__builtin_amdgcn_tensor_load_to_lds)
#define HAVE_TDM 1
#else
#define HAVE_TDM 0
#endif

// ---------------------------------------------------------------------------
// WMMA helpers (CDNA5 wave32, v_wmma_f32_16x16x32_bf16)
// ---------------------------------------------------------------------------
__device__ __forceinline__ v8f wmma_bf16(v16bf a, v16bf b, v8f c) {
    // 8 args: (neg_a, A, neg_b, B, c_mod, C, reuse_a, reuse_b)
    return __builtin_amdgcn_wmma_f32_16x16x32_bf16(false, a, false, b,
                                                   (short)0, c, false, false);
}

// A-matrix 16x32 bf16, ISA layout: lane<16 -> k {0..7,16..23}, lane>=16 -> +8.
__device__ __forceinline__ v16bf load_A_bf16(const __bf16* src, int ld, int lane) {
    int row = lane & 15;
    int kb  = (lane >> 4) * 8;
    const __bf16* p = src + (i64)row * ld + kb;
    union { v16bf v; uint4 u[2]; } r;
    r.u[0] = *(const uint4*)(p);        // k = kb .. kb+7
    r.u[1] = *(const uint4*)(p + 16);   // k = kb+16 .. kb+23
    return r.v;
}

// Same A layout but converting from f32 source in-flight.
__device__ __forceinline__ v16bf load_A_f32(const float* src, int ld, int lane) {
    int row = lane & 15;
    int kb  = (lane >> 4) * 8;
    const float* p = src + (i64)row * ld;
    v16bf a;
#pragma unroll
    for (int e = 0; e < 8; ++e) a[e]     = (__bf16)p[kb + e];
#pragma unroll
    for (int e = 0; e < 8; ++e) a[e + 8] = (__bf16)p[kb + 16 + e];
    return a;
}

// B-matrix 32x16 bf16 from Bt[m][k] storage (k contiguous per output column m).
__device__ __forceinline__ v16bf load_B_bf16(const __bf16* Bt, int ldb, int lane) {
    int m  = lane & 15;
    int kb = (lane >> 4) * 8;
    const __bf16* p = Bt + (i64)m * ldb + kb;
    union { v16bf v; uint4 u[2]; } r;
    r.u[0] = *(const uint4*)(p);
    r.u[1] = *(const uint4*)(p + 16);
    return r.v;
}

// ---------------------------------------------------------------------------
// Tensor Data Mover: DMA a 2-D bf16 tile (tile0 x tile1 elems, row pitch
// stride0 elems) from global into LDS at byte offset lds_off.
// D# packing per CDNA5 ISA.
// ---------------------------------------------------------------------------
__device__ __forceinline__ unsigned lds_offset_of(const void* p) {
    return (unsigned)(unsigned long long)
        (__attribute__((address_space(3))) const char*)p;
}

__device__ __forceinline__ void tdm_load_2d_bf16(unsigned lds_off, const void* gptr,
                                                 unsigned tile0, unsigned tile1,
                                                 unsigned tens0, unsigned tens1,
                                                 unsigned long long stride0) {
#if HAVE_TDM
    unsigned long long ga = (unsigned long long)gptr;
    u32x4 g0;
    g0.x = 1u;                                           // count=1, user mode
    g0.y = lds_off;                                      // lds_addr [63:32]
    g0.z = (unsigned)ga;                                 // global_addr [95:64]
    g0.w = (unsigned)((ga >> 32) & 0x01FFFFFFu)          // global_addr [120:96]
         | (2u << 30);                                   // type=2 ("image")
    i32x8 g1;
    g1[0] = (int)(1u << 16);                             // wg_mask=0, data_size=1 (2B)
    g1[1] = (int)((tens0 & 0xFFFFu) << 16);              // tensor_dim0[15:0] @ [63:48]
    g1[2] = (int)(((tens0 >> 16) & 0xFFFFu)              // tensor_dim0[31:16]
         |        ((tens1 & 0xFFFFu) << 16));            // tensor_dim1[15:0]
    g1[3] = (int)(((tens1 >> 16) & 0xFFFFu)              // tensor_dim1[31:16]
         |        ((tile0 & 0xFFFFu) << 16));            // tile_dim0 @ [127:112]
    g1[4] = (int)(tile1 & 0xFFFFu);                      // tile_dim1; tile_dim2=0
    g1[5] = (int)(unsigned)(stride0 & 0xFFFFFFFFu);      // dim0_stride[31:0]
    g1[6] = (int)(unsigned)((stride0 >> 32) & 0xFFFFu);  // dim0_stride[47:32]
    g1[7] = 0;                                           // dim1_stride=0 (2-D)
    i32x4 gz = {};
#if defined(__clang_major__) && (__clang_major__ >= 23)
    i32x8 gz8 = {};
    __builtin_amdgcn_tensor_load_to_lds(g0, g1, gz, gz, gz8, 0);
#else
    __builtin_amdgcn_tensor_load_to_lds(g0, g1, gz, gz, 0);
#endif
#else
    (void)lds_off; (void)gptr; (void)tile0; (void)tile1;
    (void)tens0; (void)tens1; (void)stride0;
#endif
}

// ---------------------------------------------------------------------------
// Prep kernels
// ---------------------------------------------------------------------------
__global__ void k_amask(const float* __restrict__ A, float* __restrict__ outA) {
    int idx = blockIdx.x * 256 + threadIdx.x;   // 1024 elements
    int i = idx >> 5, k = idx & 31;
    outA[idx] = (i == k) ? 0.f : A[idx];
}

// out[b][c][r] = bf16(in[b][r][c])
__global__ void k_transpose_bf16(const float* __restrict__ in, __bf16* __restrict__ out,
                                 int rows, int cols, int batch) {
    i64 per = (i64)rows * cols;
    i64 total = per * batch;
    for (i64 idx = (i64)blockIdx.x * blockDim.x + threadIdx.x; idx < total;
         idx += (i64)gridDim.x * blockDim.x) {
        i64 b = idx / per;
        i64 rem = idx - b * per;
        int c = (int)(rem / rows);
        int r = (int)(rem - (i64)c * rows);
        out[idx] = (__bf16)in[b * per + (i64)r * cols + c];
    }
}

__global__ void k_convert_bf16(const float* __restrict__ in, __bf16* __restrict__ out, i64 n) {
    for (i64 idx = (i64)blockIdx.x * blockDim.x + threadIdx.x; idx < n;
         idx += (i64)gridDim.x * blockDim.x)
        out[idx] = (__bf16)in[idx];
}

// ---------------------------------------------------------------------------
// Stage 1: weighted_sum[k,n,j] = sum_i Amask[i,k] * H_prev[i,n,j]  (bf16 out)
// Memory-bound (128 MB in / 64 MB out): VALU with A in LDS.
// ---------------------------------------------------------------------------
__global__ void k_aggregate(const float* __restrict__ Hprev, const float* __restrict__ Adag,
                            __bf16* __restrict__ WS) {
    __shared__ float As[32][32];
    int t = threadIdx.x;
    for (int idx = t; idx < 1024; idx += 256) {
        int i = idx >> 5, k = idx & 31;
        As[i][k] = (i == k) ? 0.f : Adag[idx];
    }
    __syncthreads();
    i64 col = (i64)blockIdx.x * 256 + t;   // over N*H
    float h[32];
#pragma unroll
    for (int i = 0; i < 32; ++i) h[i] = Hprev[(i64)i * NHc + col];
#pragma unroll
    for (int k = 0; k < 32; ++k) {
        float acc = 0.f;
#pragma unroll
        for (int i = 0; i < 32; ++i) acc = fmaf(As[i][k], h[i], acc);
        WS[(i64)k * NHc + col] = (__bf16)acc;
    }
}

// ---------------------------------------------------------------------------
// Stage 2: driver_emb = relu(driver @ driver_W + b)   [N,128], K=64, WMMA
// ---------------------------------------------------------------------------
__global__ void k_driver(const float* __restrict__ driver, const __bf16* __restrict__ dWt,
                         const float* __restrict__ db, float* __restrict__ emb) {
    int lane = threadIdx.x & 31, wave = threadIdx.x >> 5;
    i64 rowbase = (i64)blockIdx.x * 16;
    const float* Asrc = driver + rowbase * Dd;
    const __bf16* Bt = dWt + (i64)wave * 16 * Dd;
    v8f acc = {};
#pragma unroll
    for (int kk = 0; kk < 2; ++kk) {
        v16bf a = load_A_f32(Asrc + kk * 32, Dd, lane);
        v16bf b = load_B_bf16(Bt + kk * 32, Dd, lane);
        acc = wmma_bf16(a, b, acc);
    }
    int col = (lane & 15) + wave * 16;
    int rr  = (lane >> 4) * 8;
    float bv = db[col];
#pragma unroll
    for (int r = 0; r < 8; ++r)
        emb[(rowbase + r + rr) * Hh + col] = fmaxf(acc[r] + bv, 0.f);
}

// ---------------------------------------------------------------------------
// Stage 3 (fused struct MLP, per q, 16-row tile), TDM-staged operands:
//   TDM: A tile (4KB) + W1t[q] (32KB) -> LDS;  C1 = relu(A @ W1) -> LDS bf16
//   TDM: W2t[q] -> same LDS buffer;            Hhat = C1 @ W2 + b2 -> f32
// ---------------------------------------------------------------------------
__global__ void k_struct(const __bf16* __restrict__ WS, const __bf16* __restrict__ W1t,
                         const __bf16* __restrict__ W2t, const float* __restrict__ b1,
                         const float* __restrict__ b2, float* __restrict__ Hhat) {
    __shared__ __align__(16) __bf16 WB[Hh * Hh];   // 32KB weight buffer (W1 then W2)
    __shared__ __align__(16) __bf16 AT[16 * Hh];   // 4KB A tile
    __shared__ __align__(16) __bf16 C1[16 * Hh];   // 4KB layer-1 activations
    int q = blockIdx.y;
    i64 rowbase = (i64)blockIdx.x * 16;
    int t = threadIdx.x, lane = t & 31, wave = t >> 5;
    int col = (lane & 15) + wave * 16;
    int rr  = (lane >> 4) * 8;

    const __bf16* Ag = WS + (i64)q * NHc + rowbase * Hh;
    const __bf16* W1g = W1t + (i64)q * Hh * Hh;
    const __bf16* W2g = W2t + (i64)q * Hh * Hh;

#if HAVE_TDM
    if (wave == 0) {
        // A tile: 16x128 bf16 rows are contiguous -> one 2048-element line.
        tdm_load_2d_bf16(lds_offset_of(AT), Ag, 2048u, 1u, 2048u, 1u, 2048ull);
        // W1t[q]: 128x128 bf16, pitch 128.
        tdm_load_2d_bf16(lds_offset_of(WB), W1g, (unsigned)Hh, (unsigned)Hh,
                         (unsigned)Hh, (unsigned)Hh, (unsigned long long)Hh);
        __builtin_amdgcn_s_wait_tensorcnt((short)0);
    }
#else
    {
        const uint4* sw = (const uint4*)W1g;  uint4* dw = (uint4*)WB;
        for (int i = t; i < Hh * Hh / 8; i += 256) dw[i] = sw[i];
        const uint4* sa = (const uint4*)Ag;   uint4* da = (uint4*)AT;
        for (int i = t; i < 16 * Hh / 8; i += 256) da[i] = sa[i];
    }
#endif
    __syncthreads();

    v8f acc = {};
#pragma unroll
    for (int kk = 0; kk < 4; ++kk) {
        v16bf a = load_A_bf16(AT + kk * 32, Hh, lane);
        v16bf b = load_B_bf16(WB + (i64)wave * 16 * Hh + kk * 32, Hh, lane);
        acc = wmma_bf16(a, b, acc);
    }
    float bias1 = b1[q * Hh + col];
#pragma unroll
    for (int r = 0; r < 8; ++r)
        C1[(r + rr) * Hh + col] = (__bf16)fmaxf(acc[r] + bias1, 0.f);
    __syncthreads();   // all waves done reading WB (W1) and writing C1

#if HAVE_TDM
    if (wave == 0) {
        tdm_load_2d_bf16(lds_offset_of(WB), W2g, (unsigned)Hh, (unsigned)Hh,
                         (unsigned)Hh, (unsigned)Hh, (unsigned long long)Hh);
        __builtin_amdgcn_s_wait_tensorcnt((short)0);
    }
#else
    {
        const uint4* sw = (const uint4*)W2g;  uint4* dw = (uint4*)WB;
        for (int i = t; i < Hh * Hh / 8; i += 256) dw[i] = sw[i];
    }
#endif
    __syncthreads();

    v8f acc2 = {};
#pragma unroll
    for (int kk = 0; kk < 4; ++kk) {
        v16bf a = load_A_bf16(C1 + kk * 32, Hh, lane);
        v16bf b = load_B_bf16(WB + (i64)wave * 16 * Hh + kk * 32, Hh, lane);
        acc2 = wmma_bf16(a, b, acc2);
    }
    float bias2 = b2[q * Hh + col];
    float* Od = Hhat + (i64)q * NHc + rowbase * Hh;
#pragma unroll
    for (int r = 0; r < 8; ++r)
        Od[(i64)(r + rr) * Hh + col] = acc2[r] + bias2;
}

// ---------------------------------------------------------------------------
// Stage 4 (fused GRU, per q, 16-row tile):
//   gi = (emb + var_embed[q]) @ Wih[q]^T ; gh = Hhat[q] @ Whh[q]^T  (WMMA)
//   then sigmoid/tanh gates + blend -> H_next
// ---------------------------------------------------------------------------
__global__ void k_gru(const float* __restrict__ emb, const float* __restrict__ vemb,
                      const float* __restrict__ Hhat, const __bf16* __restrict__ Wih,
                      const __bf16* __restrict__ Whh, const float* __restrict__ bih,
                      const float* __restrict__ bhh, float* __restrict__ Hnext) {
    __shared__ __align__(16) float GI[16 * THc];
    __shared__ __align__(16) float GH[16 * THc];
    __shared__ __align__(16) __bf16 Ai[16 * Hh];
    __shared__ __align__(16) __bf16 Ah[16 * Hh];
    int q = blockIdx.y;
    i64 rowbase = (i64)blockIdx.x * 16;
    int t = threadIdx.x, lane = t & 31, wave = t >> 5;

    for (int idx = t; idx < 16 * Hh; idx += 256) {
        int r = idx >> 7, c = idx & 127;
        Ai[idx] = (__bf16)(emb[(rowbase + r) * Hh + c] + vemb[q * Hh + c]);
        Ah[idx] = (__bf16)Hhat[(i64)q * NHc + (rowbase + r) * Hh + c];
    }
    __syncthreads();

#pragma unroll
    for (int tt = 0; tt < 6; ++tt) {
        int tile = wave * 6 + tt;
        int isH  = (tile >= 24);
        int c    = isH ? (tile - 24) : tile;
        const __bf16* Asrc = isH ? Ah : Ai;
        const __bf16* Bt   = (isH ? Whh : Wih) + (i64)q * THc * Hh + (i64)c * 16 * Hh;
        v8f acc = {};
#pragma unroll
        for (int kk = 0; kk < 4; ++kk) {
            v16bf a = load_A_bf16(Asrc + kk * 32, Hh, lane);
            v16bf b = load_B_bf16(Bt + kk * 32, Hh, lane);
            acc = wmma_bf16(a, b, acc);
        }
        float* G = isH ? GH : GI;
        int col  = (lane & 15) + c * 16;
        int rr   = (lane >> 4) * 8;
        float bv = (isH ? bhh : bih)[q * THc + col];
#pragma unroll
        for (int r = 0; r < 8; ++r) G[(r + rr) * THc + col] = acc[r] + bv;
    }
    __syncthreads();

    for (int idx = t; idx < 16 * Hh; idx += 256) {
        int r = idx >> 7, c = idx & 127;
        float ir = GI[r * THc + c],        hr = GH[r * THc + c];
        float iu = GI[r * THc + 128 + c],  hu = GH[r * THc + 128 + c];
        float in_ = GI[r * THc + 256 + c], hn = GH[r * THc + 256 + c];
        float rg = 1.f / (1.f + __expf(-(ir + hr)));
        float ug = 1.f / (1.f + __expf(-(iu + hu)));
        float ng = tanhf(in_ + rg * hn);
        i64 o = (i64)q * NHc + (rowbase + r) * Hh + c;
        float hh = Hhat[o];
        Hnext[o] = (1.f - ug) * ng + ug * hh;
    }
}

// ---------------------------------------------------------------------------
// Stage 5: reconstruction [N, Q*H] @ recon_W [Q*H, R] + b -> [N, 64]
// ---------------------------------------------------------------------------
__global__ void k_recon(const float* __restrict__ Hprev, const __bf16* __restrict__ rWt,
                        const float* __restrict__ rb, float* __restrict__ outR) {
    int lane = threadIdx.x & 31, wave = threadIdx.x >> 5;
    i64 rowbase = (i64)blockIdx.x * 16;
    const __bf16* Bt = rWt + (i64)wave * 16 * (Qv * Hh);
    v8f acc = {};
    for (int kk = 0; kk < 128; ++kk) {
        int ks = kk * 32;
        const float* Asrc = Hprev + (i64)(ks >> 7) * NHc + rowbase * Hh + (ks & 127);
        __builtin_prefetch((const void*)(Bt + (i64)(kk + 1) * 32), 0, 0); // global_prefetch_b8
        v16bf a = load_A_f32(Asrc, Hh, lane);
        v16bf b = load_B_bf16(Bt + ks, Qv * Hh, lane);
        acc = wmma_bf16(a, b, acc);
    }
    int col = (lane & 15) + wave * 16;
    int rr  = (lane >> 4) * 8;
    float bv = rb[col];
#pragma unroll
    for (int r = 0; r < 8; ++r)
        outR[(rowbase + r + rr) * Rr + col] = acc[r] + bv;
}

// ---------------------------------------------------------------------------
// Host launcher
// ---------------------------------------------------------------------------
extern "C" void kernel_launch(void* const* d_in, const int* in_sizes, int n_in,
                              void* d_out, int out_size, void* d_ws, size_t ws_size,
                              hipStream_t stream) {
    (void)in_sizes; (void)n_in; (void)out_size; (void)ws_size;
    const float* Hprev = (const float*)d_in[0];
    const float* driver = (const float*)d_in[1];
    const float* Adag  = (const float*)d_in[2];
    const float* W1    = (const float*)d_in[3];
    const float* b1    = (const float*)d_in[4];
    const float* W2    = (const float*)d_in[5];
    const float* b2    = (const float*)d_in[6];
    const float* dW    = (const float*)d_in[7];
    const float* db    = (const float*)d_in[8];
    const float* Wih   = (const float*)d_in[9];
    const float* Whh   = (const float*)d_in[10];
    const float* bih   = (const float*)d_in[11];
    const float* bhh   = (const float*)d_in[12];
    const float* vemb  = (const float*)d_in[13];
    const float* rW    = (const float*)d_in[14];
    const float* rb    = (const float*)d_in[15];

    float* outH = (float*)d_out;                 // [Q,N,H]
    float* outR = outH + (i64)Qv * Nn * Hh;      // [N,R]
    float* outA = outR + (i64)Nn * Rr;           // [Q,Q]

    char* ws = (char*)d_ws;
    size_t off = 0;
    auto take = [&](size_t bytes) -> char* {
        char* p = ws + off;
        off = (off + bytes + 255) & ~(size_t)255;
        return p;
    };
    __bf16* wsWS  = (__bf16*)take((size_t)Qv * NHc * 2);       // weighted_sum bf16
    float*  wsHh  = (float*) take((size_t)Qv * NHc * 4);       // H_hat f32
    float*  wsEmb = (float*) take((size_t)NHc * 4);            // driver_emb f32
    __bf16* W1t   = (__bf16*)take((size_t)Qv * Hh * Hh * 2);
    __bf16* W2t   = (__bf16*)take((size_t)Qv * Hh * Hh * 2);
    __bf16* Wiht  = (__bf16*)take((size_t)Qv * THc * Hh * 2);
    __bf16* Whht  = (__bf16*)take((size_t)Qv * THc * Hh * 2);
    __bf16* dWt   = (__bf16*)take((size_t)Hh * Dd * 2);
    __bf16* rWt   = (__bf16*)take((size_t)Rr * Qv * Hh * 2);

    k_amask<<<4, 256, 0, stream>>>(Adag, outA);
    k_transpose_bf16<<<256, 256, 0, stream>>>(W1, W1t, Hh, Hh, Qv);
    k_transpose_bf16<<<256, 256, 0, stream>>>(W2, W2t, Hh, Hh, Qv);
    k_transpose_bf16<<<8, 256, 0, stream>>>(dW, dWt, Dd, Hh, 1);
    k_transpose_bf16<<<128, 256, 0, stream>>>(rW, rWt, Qv * Hh, Rr, 1);
    k_convert_bf16<<<512, 256, 0, stream>>>(Wih, Wiht, (i64)Qv * THc * Hh);
    k_convert_bf16<<<512, 256, 0, stream>>>(Whh, Whht, (i64)Qv * THc * Hh);

    k_aggregate<<<NHc / 256, 256, 0, stream>>>(Hprev, Adag, wsWS);
    k_driver<<<Nn / 16, 256, 0, stream>>>(driver, dWt, db, wsEmb);
    k_struct<<<dim3(Nn / 16, Qv), 256, 0, stream>>>(wsWS, W1t, W2t, b1, b2, wsHh);
    k_gru<<<dim3(Nn / 16, Qv), 256, 0, stream>>>(wsEmb, vemb, wsHh, Wiht, Whht, bih, bhh, outH);
    k_recon<<<Nn / 16, 128, 0, stream>>>(Hprev, rWt, rb, outR);
}
